// TimeMixing_62801011802494
// MI455X (gfx1250) — matrix operand order repaired and checked by reading, not compiled, not run on hardware
//
#include <hip/hip_runtime.h>
#include <hip/hip_bf16.h>
#include <stdint.h>

#define BDIM 8
#define TDIM 2048
#define CDIM 1024
#define NROWS (BDIM * TDIM)          // 16384 token rows
#define BTC ((size_t)BDIM * TDIM * CDIM)

typedef __attribute__((ext_vector_type(16))) __bf16 v16bf;
typedef __attribute__((ext_vector_type(8)))  float  v8f;
typedef __attribute__((ext_vector_type(4)))  unsigned int u32x4;
typedef __attribute__((ext_vector_type(4)))  int          i32x4;
typedef __attribute__((ext_vector_type(8)))  int          i32x8;

struct U32x8 { uint4 lo; uint4 hi; };
union BfPack { U32x8 u; v16bf v; };

static __device__ __forceinline__ unsigned short f2bf(float f) {
    union { float f; uint32_t u; } x; x.f = f;
    uint32_t r = x.u + 0x7FFFu + ((x.u >> 16) & 1u);   // round-to-nearest-even
    return (unsigned short)(r >> 16);
}
static __device__ __forceinline__ float bf2f(unsigned short h) {
    union { uint32_t u; float f; } x; x.u = ((uint32_t)h) << 16;
    return x.f;
}

#if __has_builtin(__builtin_amdgcn_tensor_load_to_lds) && __has_builtin(__builtin_amdgcn_s_wait_tensorcnt)
#define USE_TDM 1
#else
#define USE_TDM 0
#endif

// ---------------------------------------------------------------------------
// 1) Token shift + time-mix, fp32 -> bf16. Also emits last-token output.
// ---------------------------------------------------------------------------
__global__ __launch_bounds__(256)
void prep_kernel(const float* __restrict__ x, const float* __restrict__ last_x,
                 const float* __restrict__ tmk, const float* __restrict__ tmv,
                 const float* __restrict__ tmr,
                 unsigned short* __restrict__ xk, unsigned short* __restrict__ xv,
                 unsigned short* __restrict__ xr, float* __restrict__ lastx_out) {
    size_t idx = (size_t)blockIdx.x * blockDim.x + threadIdx.x;
    if (idx >= BTC) return;
    int c = (int)(idx & (CDIM - 1));
    int t = (int)((idx >> 10) & (TDIM - 1));
    int b = (int)(idx >> 21);
    float xc = x[idx];
    float xp = (t == 0) ? last_x[(size_t)b * CDIM + c] : x[idx - CDIM];
    float mk = tmk[c], mv = tmv[c], mr = tmr[c];
    xk[idx] = f2bf(xc * mk + xp * (1.0f - mk));
    xv[idx] = f2bf(xc * mv + xp * (1.0f - mv));
    xr[idx] = f2bf(xc * mr + xp * (1.0f - mr));
    if (t == TDIM - 1) lastx_out[(size_t)b * CDIM + c] = xc;
}

// ---------------------------------------------------------------------------
// 2) Weight transpose + bf16 convert: Wt[d*C + c] = bf16(W[c*C + d])
// ---------------------------------------------------------------------------
__global__ __launch_bounds__(256)
void wtconv_kernel(const float* __restrict__ W, unsigned short* __restrict__ Wt) {
    size_t idx = (size_t)blockIdx.x * blockDim.x + threadIdx.x;   // over C*C
    if (idx >= (size_t)CDIM * CDIM) return;
    int c = (int)(idx & (CDIM - 1));
    int d = (int)(idx >> 10);
    Wt[idx] = f2bf(W[(size_t)c * CDIM + d]);
}

// ---------------------------------------------------------------------------
// 3) bf16 WMMA GEMM:  C[N,D] = A[N,C] * Wt[D,C]^T   (Wt is pre-transposed)
// ---------------------------------------------------------------------------
#if USE_TDM
// Issue a 2D TDM tile load: tile (tile_d1 rows x tile_d0 elems, bf16) from a
// row-major tensor with row stride stride0 into LDS at lds_off (packed rows).
static __device__ __forceinline__ void tdm_load_2d(uint32_t lds_off_bytes,
                                                   const void* gptr,
                                                   uint32_t tile_d0, uint32_t tile_d1,
                                                   uint32_t tensor_d0, uint32_t tensor_d1,
                                                   uint32_t stride0) {
    uint64_t ga = (uint64_t)(uintptr_t)gptr;
    u32x4 g0;
    g0[0] = 1u;                                              // count=1, user descriptor
    g0[1] = lds_off_bytes;                                   // lds_addr (bytes)
    g0[2] = (uint32_t)ga;                                    // global_addr[31:0]
    g0[3] = (uint32_t)((ga >> 32) & 0x1FFFFFFu) | (2u << 30); // addr[56:32] | type=2
    i32x8 g1;
    g1[0] = (int)(1u << 16);                                 // data_size=1 (2 bytes/elem)
    g1[1] = (int)((tensor_d0 & 0xFFFFu) << 16);              // tensor_dim0[15:0] (abar=0)
    g1[2] = (int)(((tensor_d1 & 0xFFFFu) << 16) | (tensor_d0 >> 16));
    g1[3] = (int)((tile_d0 << 16) | (tensor_d1 >> 16));      // tile_dim0 | tensor_dim1 hi
    g1[4] = (int)tile_d1;                                    // tile_dim1 (tile_dim2=0)
    g1[5] = (int)stride0;                                    // tensor_dim0_stride[31:0]
    g1[6] = 0;                                               // stride0 hi / dim1_stride lo
    g1[7] = 0;                                               // dim1_stride hi
    i32x4 z4 = {0, 0, 0, 0};
#if defined(__clang_major__) && (__clang_major__ >= 23)
    i32x8 z8 = {0, 0, 0, 0, 0, 0, 0, 0};
    __builtin_amdgcn_tensor_load_to_lds(g0, g1, z4, z4, z8, 0);
#else
    __builtin_amdgcn_tensor_load_to_lds(g0, g1, z4, z4, 0);
#endif
}

// Block = 256 threads = 8 waves; block tile 128x64 (waves 4x2, 32x32 each).
// K-chunk 64, double-buffered TDM into LDS: A 128x64 bf16 (16KB) + B 64x64 (8KB).
__global__ __launch_bounds__(256)
void gemm_bf16_kernel(const unsigned short* __restrict__ A,
                      const unsigned short* __restrict__ Bt,
                      float* __restrict__ Cf,
                      unsigned short* __restrict__ Cb,
                      int mode) {
    extern __shared__ unsigned short smem[];
    constexpr int KC = 64;
    constexpr int AELEMS = 128 * KC;       // 8192 elems
    constexpr int BELEMS = 64 * KC;        // 4096 elems
    constexpr int STAGE  = AELEMS + BELEMS;

    const int lane = threadIdx.x & 31;
    const int wave = threadIdx.x >> 5;
    const int nb   = CDIM / 64;            // 16 col-blocks
    const int row0 = (blockIdx.x / nb) * 128;
    const int col0 = (blockIdx.x % nb) * 64;
    const int wrow = (wave >> 1) * 32;
    const int wcol = (wave & 1) * 32;
    const int m  = lane & 15;
    const int kh = lane >> 4;

    const unsigned short* aBase = A  + (size_t)row0 * CDIM;
    const unsigned short* bBase = Bt + (size_t)col0 * CDIM;

    v8f acc00 = {}, acc01 = {}, acc10 = {}, acc11 = {};

    const int nstages = CDIM / KC;         // 16
    if (wave == 0) {
        tdm_load_2d(0u,          aBase, KC, 128, CDIM, NROWS, CDIM);
        tdm_load_2d(AELEMS * 2u, bBase, KC,  64, CDIM, CDIM,  CDIM);
    }
    for (int s = 0; s < nstages; ++s) {
        if (wave == 0) {
            if (s + 1 < nstages) {
                unsigned off = ((s + 1) & 1) ? (unsigned)(STAGE * 2) : 0u;
                tdm_load_2d(off,              aBase + (size_t)(s + 1) * KC, KC, 128, CDIM, NROWS, CDIM);
                tdm_load_2d(off + AELEMS * 2, bBase + (size_t)(s + 1) * KC, KC,  64, CDIM, CDIM,  CDIM);
                __builtin_amdgcn_s_wait_tensorcnt(2);   // stage s (oldest 2 ops) done
            } else {
                __builtin_amdgcn_s_wait_tensorcnt(0);
            }
        }
        __syncthreads();                    // publish stage s LDS data
        const unsigned short* sA = smem + (s & 1) * STAGE;
        const unsigned short* sB = sA + AELEMS;
#pragma unroll
        for (int kk = 0; kk < KC; kk += 32) {
            BfPack a0, a1, b0, b1;
            // A operand, this code's k order (same as B): 16 contiguous K at kh*16, row = lane&15
            const unsigned short* a0p = sA + (wrow + m) * KC + kk + kh * 16;
            const unsigned short* a1p = a0p + 16 * KC;
            // B operand, this code's k order (same as A): 16 contiguous K at kh*16, col = lane&15
            const unsigned short* b0p = sB + (wcol + m) * KC + kk + kh * 16;
            const unsigned short* b1p = b0p + 16 * KC;
            a0.u.lo = *(const uint4*)(a0p);      a0.u.hi = *(const uint4*)(a0p + 8);
            a1.u.lo = *(const uint4*)(a1p);      a1.u.hi = *(const uint4*)(a1p + 8);
            b0.u.lo = *(const uint4*)(b0p);      b0.u.hi = *(const uint4*)(b0p + 8);
            b1.u.lo = *(const uint4*)(b1p);      b1.u.hi = *(const uint4*)(b1p + 8);
            acc00 = __builtin_amdgcn_wmma_f32_16x16x32_bf16(false, a0.v, false, b0.v, (short)0, acc00, false, false);
            acc01 = __builtin_amdgcn_wmma_f32_16x16x32_bf16(false, a0.v, false, b1.v, (short)0, acc01, false, false);
            acc10 = __builtin_amdgcn_wmma_f32_16x16x32_bf16(false, a1.v, false, b0.v, (short)0, acc10, false, false);
            acc11 = __builtin_amdgcn_wmma_f32_16x16x32_bf16(false, a1.v, false, b1.v, (short)0, acc11, false, false);
        }
        __syncthreads();                    // stage s fully consumed before reuse
    }

    // C/D layout: VGPR r -> row = r + (lane>>4)*8, col = lane&15
    const int gr = row0 + wrow + kh * 8;
    const int gc = col0 + wcol + m;
    if (mode == 0) {
        for (int r = 0; r < 8; ++r) {
            size_t ro = (size_t)(gr + r) * CDIM;
            Cf[ro + gc]                         = acc00[r];
            Cf[ro + gc + 16]                    = acc01[r];
            Cf[ro + (size_t)16 * CDIM + gc]      = acc10[r];
            Cf[ro + (size_t)16 * CDIM + gc + 16] = acc11[r];
        }
    } else {
        for (int r = 0; r < 8; ++r) {
            size_t ro = (size_t)(gr + r) * CDIM;
            Cb[ro + gc]                         = f2bf(1.0f / (1.0f + __expf(-acc00[r])));
            Cb[ro + gc + 16]                    = f2bf(1.0f / (1.0f + __expf(-acc01[r])));
            Cb[ro + (size_t)16 * CDIM + gc]      = f2bf(1.0f / (1.0f + __expf(-acc10[r])));
            Cb[ro + (size_t)16 * CDIM + gc + 16] = f2bf(1.0f / (1.0f + __expf(-acc11[r])));
        }
    }
}
#else  // !USE_TDM: direct-global fallback (round-1 scheme, 32x32 tile per wave)
__global__ __launch_bounds__(256)
void gemm_bf16_kernel(const unsigned short* __restrict__ A,
                      const unsigned short* __restrict__ Bt,
                      float* __restrict__ Cf,
                      unsigned short* __restrict__ Cb,
                      int mode) {
    const int lane = threadIdx.x & 31;
    const int wave = threadIdx.x >> 5;
    const int tile = blockIdx.x * 8 + wave;
    const int NT   = CDIM / 32;
    const int row0 = (tile / NT) * 32;
    const int col0 = (tile % NT) * 32;
    const int m  = lane & 15;
    const int kh = lane >> 4;

    v8f acc00 = {}, acc01 = {}, acc10 = {}, acc11 = {};
    const unsigned short* a0p = A  + (size_t)(row0 + m) * CDIM + kh * 16;
    const unsigned short* a1p = a0p + (size_t)16 * CDIM;
    const unsigned short* b0p = Bt + (size_t)(col0 + m) * CDIM + kh * 16;
    const unsigned short* b1p = b0p + (size_t)16 * CDIM;

    for (int kk = 0; kk < CDIM; kk += 32) {
        BfPack a0, a1, b0, b1;
        a0.u.lo = *(const uint4*)(a0p + kk);  a0.u.hi = *(const uint4*)(a0p + kk + 8);
        a1.u.lo = *(const uint4*)(a1p + kk);  a1.u.hi = *(const uint4*)(a1p + kk + 8);
        b0.u.lo = *(const uint4*)(b0p + kk);  b0.u.hi = *(const uint4*)(b0p + kk + 8);
        b1.u.lo = *(const uint4*)(b1p + kk);  b1.u.hi = *(const uint4*)(b1p + kk + 8);
        acc00 = __builtin_amdgcn_wmma_f32_16x16x32_bf16(false, a0.v, false, b0.v, (short)0, acc00, false, false);
        acc01 = __builtin_amdgcn_wmma_f32_16x16x32_bf16(false, a0.v, false, b1.v, (short)0, acc01, false, false);
        acc10 = __builtin_amdgcn_wmma_f32_16x16x32_bf16(false, a1.v, false, b0.v, (short)0, acc10, false, false);
        acc11 = __builtin_amdgcn_wmma_f32_16x16x32_bf16(false, a1.v, false, b1.v, (short)0, acc11, false, false);
    }
    const int mrow = kh * 8;
    if (mode == 0) {
        for (int r = 0; r < 8; ++r) {
            size_t ro = (size_t)(row0 + mrow + r) * CDIM;
            Cf[ro + col0 + m]                         = acc00[r];
            Cf[ro + col0 + 16 + m]                    = acc01[r];
            Cf[ro + (size_t)16 * CDIM + col0 + m]      = acc10[r];
            Cf[ro + (size_t)16 * CDIM + col0 + 16 + m] = acc11[r];
        }
    } else {
        for (int r = 0; r < 8; ++r) {
            size_t ro = (size_t)(row0 + mrow + r) * CDIM;
            Cb[ro + col0 + m]                         = f2bf(1.0f / (1.0f + __expf(-acc00[r])));
            Cb[ro + col0 + 16 + m]                    = f2bf(1.0f / (1.0f + __expf(-acc01[r])));
            Cb[ro + (size_t)16 * CDIM + col0 + m]      = f2bf(1.0f / (1.0f + __expf(-acc10[r])));
            Cb[ro + (size_t)16 * CDIM + col0 + 16 + m] = f2bf(1.0f / (1.0f + __expf(-acc11[r])));
        }
    }
}
#endif

// ---------------------------------------------------------------------------
// 4) WKV serial scan: one thread per (b,c) channel, fp32 state.
// ---------------------------------------------------------------------------
__global__ __launch_bounds__(256)
void wkv_scan_kernel(const float* __restrict__ k, const float* __restrict__ v,
                     const unsigned short* __restrict__ rsig,
                     const float* __restrict__ aa_in, const float* __restrict__ bb_in,
                     const float* __restrict__ pp_in,
                     const float* __restrict__ time_first, const float* __restrict__ time_decay,
                     unsigned short* __restrict__ rwkv,
                     float* __restrict__ aa_out, float* __restrict__ bb_out,
                     float* __restrict__ pp_out) {
    int gid = blockIdx.x * blockDim.x + threadIdx.x;    // [0, B*C)
    if (gid >= BDIM * CDIM) return;
    int c = gid & (CDIM - 1);
    int b = gid >> 10;
    float aa = aa_in[gid], bb = bb_in[gid], pp = pp_in[gid];
    const float tf = time_first[c];
    const float td = time_decay[c];
    size_t base = (size_t)b * TDIM * CDIM + c;
    for (int t = 0; t < TDIM; ++t) {
        size_t off = base + (size_t)t * CDIM;
        float kt = k[off];
        float vt = v[off];
        float ww = tf + kt;
        float qq = fmaxf(pp, ww);
        float e1 = __expf(pp - qq);
        float e2 = __expf(ww - qq);
        float wkv = (e1 * aa + e2 * vt) / (e1 * bb + e2);
        rwkv[off] = f2bf(bf2f(rsig[off]) * wkv);
        float ww2 = pp + td;
        float qq2 = fmaxf(ww2, kt);
        float e1b = __expf(ww2 - qq2);
        float e2b = __expf(kt - qq2);
        aa = e1b * aa + e2b * vt;
        bb = e1b * bb + e2b;
        pp = qq2;
    }
    aa_out[gid] = aa;
    bb_out[gid] = bb;
    pp_out[gid] = pp;
}

// ---------------------------------------------------------------------------
// Host-side launch sequence
// ---------------------------------------------------------------------------
extern "C" void kernel_launch(void* const* d_in, const int* in_sizes, int n_in,
                              void* d_out, int out_size, void* d_ws, size_t ws_size,
                              hipStream_t stream) {
    const float* x          = (const float*)d_in[0];
    const float* last_x     = (const float*)d_in[1];
    const float* aa_in      = (const float*)d_in[2];
    const float* bb_in      = (const float*)d_in[3];
    const float* pp_in      = (const float*)d_in[4];
    const float* time_mix_k = (const float*)d_in[5];
    const float* time_mix_v = (const float*)d_in[6];
    const float* time_mix_r = (const float*)d_in[7];
    const float* time_first = (const float*)d_in[8];
    const float* time_decay = (const float*)d_in[9];
    const float* Wk         = (const float*)d_in[10];
    const float* Wv         = (const float*)d_in[11];
    const float* Wr         = (const float*)d_in[12];
    const float* Wo         = (const float*)d_in[13];

    // Output segments: out[B,T,C] | last_x[B,C] | aa | bb | pp
    float* out_y    = (float*)d_out;
    float* out_last = out_y + BTC;
    float* out_aa   = out_last + (size_t)BDIM * CDIM;
    float* out_bb   = out_aa + (size_t)BDIM * CDIM;
    float* out_pp   = out_bb + (size_t)BDIM * CDIM;

    // Workspace layout (bf16 stored as ushort)
    unsigned short* xk   = (unsigned short*)d_ws;              // BTC
    unsigned short* xv   = xk + BTC;                           // BTC
    unsigned short* xr   = xv + BTC;                           // BTC
    unsigned short* WkT  = xr + BTC;                           // C*C each
    unsigned short* WvT  = WkT + (size_t)CDIM * CDIM;
    unsigned short* WrT  = WvT + (size_t)CDIM * CDIM;
    unsigned short* WoT  = WrT + (size_t)CDIM * CDIM;
    float* kbuf          = (float*)(WoT + (size_t)CDIM * CDIM); // BTC f32
    float* vbuf          = kbuf + BTC;                          // BTC f32
    unsigned short* rsig = xv;   // alias: xv dead after v GEMM (stream-ordered)
    unsigned short* rwkv = xk;   // alias: xk dead after k GEMM

    // 1) prep
    prep_kernel<<<dim3((unsigned)(BTC / 256)), 256, 0, stream>>>(
        x, last_x, time_mix_k, time_mix_v, time_mix_r, xk, xv, xr, out_last);

    // 2) weight transpose + convert
    unsigned wblocks = (unsigned)(((size_t)CDIM * CDIM) / 256);
    wtconv_kernel<<<wblocks, 256, 0, stream>>>(Wk, WkT);
    wtconv_kernel<<<wblocks, 256, 0, stream>>>(Wv, WvT);
    wtconv_kernel<<<wblocks, 256, 0, stream>>>(Wr, WrT);
    wtconv_kernel<<<wblocks, 256, 0, stream>>>(Wo, WoT);

    // 3) GEMMs: 2048 blocks x 256 threads; 48KB dynamic LDS for TDM double-buffer
    const unsigned gblocks = 2048;
    const size_t   gsmem   = 2u * (128 * 64 + 64 * 64) * sizeof(unsigned short); // 49152
    gemm_bf16_kernel<<<gblocks, 256, gsmem, stream>>>(xk, WkT, kbuf, nullptr, 0);
    gemm_bf16_kernel<<<gblocks, 256, gsmem, stream>>>(xv, WvT, vbuf, nullptr, 0);
    gemm_bf16_kernel<<<gblocks, 256, gsmem, stream>>>(xr, WrT, nullptr, rsig, 1);

    // 4) serial WKV scan: 8192 channels
    wkv_scan_kernel<<<(BDIM * CDIM) / 256, 256, 0, stream>>>(
        kbuf, vbuf, rsig, aa_in, bb_in, pp_in, time_first, time_decay,
        rwkv, out_aa, out_bb, out_pp);

    // 5) output projection -> d_out (f32)
    gemm_bf16_kernel<<<gblocks, 256, gsmem, stream>>>(rwkv, WoT, out_y, nullptr, 0);
}
